// EdgeUpdate_20607253086908
// MI455X (gfx1250) — matrix operand order repaired and checked by reading, not compiled, not run on hardware
//
#include <hip/hip_runtime.h>
#include <hip/hip_bf16.h>

typedef __attribute__((ext_vector_type(16))) __bf16       v16bf;
typedef __attribute__((ext_vector_type(8)))  float        v8f;
typedef __attribute__((ext_vector_type(4)))  unsigned int u32x4;
typedef __attribute__((ext_vector_type(8)))  unsigned int u32x8;

#define N_NODES   16384
#define E_EDGES   32768
#define H_DIM     64
#define RES_DIMC  56
#define W_INC     80      // d dimension
#define H1_DIM    128     // 2*H

// LDS layout (manual overlay, total 59904 B < 64 KB):
//  [0,      16384) efS     : 64x64 f32 edge-feature / x tile
//  [16384,  26624) featsS  : 64x80 bf16
//  [26624,  43008) h1S     : 64x128 bf16  == TDM buffer 1 (reused after frags)
//  [43008,  59392) w2buf0  : 64x128 bf16  == TDM buffer 0 (also b2 in epilogue)
//  [59392,  59904) muS/rsS
#define OFF_EFS    0
#define OFF_FEATS  16384
#define OFF_H1     26624
#define OFF_W2B0   43008
#define OFF_MU     59392
#define SMEM_BYTES 59904

// ---------------------------------------------------------------------------
// Kernel 1: W2 (128 x 5120 f32, col = d*64+h) -> W2T[d][h][k1] bf16
// ---------------------------------------------------------------------------
__global__ __launch_bounds__(256)
void w2t_kernel(const float* __restrict__ W2, __bf16* __restrict__ W2T) {
    int idx = blockIdx.x * 256 + threadIdx.x;      // 0 .. 655359
    if (idx < W_INC * H_DIM * H1_DIM) {
        int k1 = idx & 127;
        int h  = (idx >> 7) & 63;
        int d  = idx >> 13;
        W2T[idx] = (__bf16)W2[k1 * (W_INC * H_DIM) + d * H_DIM + h];
    }
}

// ---------------------------------------------------------------------------
// TDM: issue tensor_load_to_lds of one contiguous 16KB block (4096 dwords).
// D# per CDNA5 ISA: group0 = {count, lds_addr,
// global_addr(57b), type=2}; group1 = {mask=0, data_size=4B, dims}.
// 2-group form (VADDR2/3 NULL) is valid for tensors up to 2D.
// ---------------------------------------------------------------------------
__device__ __forceinline__
void tdm_load_16k(unsigned lds_addr, const void* gsrc) {
    unsigned long long ga = (unsigned long long)gsrc;
    u32x4 g0;
    g0[0] = 1u;                                    // count=1, is_restore=0
    g0[1] = lds_addr;                              // LDS byte address
    g0[2] = (unsigned)ga;                          // global_addr[31:0]
    g0[3] = (unsigned)(ga >> 32) | (2u << 30);     // global_addr[56:32] | type=2
    u32x8 g1;
    g1[0] = (2u << 16);                            // wg_mask=0, data_size=2 (4B)
    g1[1] = (4096u & 0xffffu) << 16;               // tensor_dim0[15:0] @bits[63:48]
    g1[2] = 0u;                                    // tensor_dim0[31:16], tensor_dim1 lo
    g1[3] = (4096u << 16);                         // tile_dim0=4096 @bits[127:112]
    g1[4] = 0u;                                    // tile_dim1=0, tile_dim2=0 (unused)
    g1[5] = 4096u;                                 // tensor_dim0_stride lo
    g1[6] = 0u;
    g1[7] = 0u;
    asm volatile("tensor_load_to_lds %0, %1" :: "s"(g0), "s"(g1) : "memory");
}

// ---------------------------------------------------------------------------
// Kernel 2: fused edge update. 1 block = 128 threads = 4 waves = 64 edges.
// tp = P(Ex10240) @ W2T'(10240x64) via v_wmma_f32_16x16x32_bf16;
// W2T d-blocks streamed into LDS by the Tensor Data Mover, double-buffered.
// ---------------------------------------------------------------------------
__global__ __launch_bounds__(128)
void edge_main(const float* __restrict__ res,        // N x 56
               const float* __restrict__ ef,         // E x 64
               const float* __restrict__ esh,        // E x 4
               const long long* __restrict__ eidx,   // 2 x E (int64)
               const float* __restrict__ W1,         // 64 x 128
               const float* __restrict__ b1,         // 128
               const float* __restrict__ b2,         // 5120
               const float* __restrict__ gamma_,     // 64
               const float* __restrict__ beta_,      // 64
               const __bf16* __restrict__ W2T,       // 80 x 64 x 128 bf16
               float* __restrict__ out)              // E x 64
{
    __shared__ __align__(16) unsigned char smem[SMEM_BYTES];
    float*  efS    = (float*) (smem + OFF_EFS);
    __bf16* featsS = (__bf16*)(smem + OFF_FEATS);
    __bf16* h1S    = (__bf16*)(smem + OFF_H1);     // == w2 buffer 1
    __bf16* w2b0   = (__bf16*)(smem + OFF_W2B0);   // == b2 staging in epilogue
    __bf16* w2b1   = (__bf16*)(smem + OFF_H1);
    float*  muS    = (float*) (smem + OFF_MU);
    float*  rsS    = (float*) (smem + OFF_MU + 256);

    const int t  = threadIdx.x;
    const int e0 = blockIdx.x * 64;

    // LDS byte offsets for TDM descriptors (flat LDS aperture: low 32 bits).
    const unsigned lds_b0 = (unsigned)(unsigned long long)(uintptr_t)w2b0;
    const unsigned lds_b1 = (unsigned)(unsigned long long)(uintptr_t)w2b1;

    // pre-issue TDM load of d=0 block; overlaps all of phase 1
    if (t < 32)
        tdm_load_16k(lds_b0, W2T);

    // prefetch epilogue b2 into cache
    __builtin_prefetch(b2 + t * 40, 0, 2);

    // ---------------- phase 1a: edge-feature tile ----------------
    #pragma unroll
    for (int j = 0; j < 32; ++j) {                   // 4096 f32, coalesced
        int idx = t + 128 * j;
        efS[idx] = ef[e0 * 64 + idx];
    }

    // ---------------- phase 1b: gather + feats -------------------
    {
        const float inv_sqrt3 = 0.57735026918962576f;
        int el    = t >> 1;          // local edge 0..63
        int which = t & 1;           // 1 = src (edge_index[1]), 0 = dst
        long long node = eidx[(long long)which * E_EDGES + (e0 + el)];
        const float* r = res + node * RES_DIMC;
        float sh0 = esh[(e0 + el) * 4 + 0];
        float s1x = esh[(e0 + el) * 4 + 1];
        float s1y = esh[(e0 + el) * 4 + 2];
        float s1z = esh[(e0 + el) * 4 + 3];
        int sbase = which ? 0 : 32;      // feats[0:32]=src*sh0, [32:64]=dst*sh0
        int vbase = which ? 64 : 72;     // feats[64:72]=src dots, [72:80]=dst
        #pragma unroll 8
        for (int i = 0; i < 32; ++i)
            featsS[el * 80 + sbase + i] = (__bf16)(r[i] * sh0);
        #pragma unroll
        for (int j = 0; j < 8; ++j) {
            float dp = r[32 + 3 * j] * s1x + r[33 + 3 * j] * s1y + r[34 + 3 * j] * s1z;
            featsS[el * 80 + vbase + j] = (__bf16)(dp * inv_sqrt3);
        }
    }
    __syncthreads();

    // ---------------- phase 1c: h1 = relu(ef @ W1 + b1) ----------
    {
        float bcol = b1[t];                  // thread t owns h1 column t
        for (int e = 0; e < 64; ++e) {
            float acc = bcol;
            #pragma unroll 8
            for (int i = 0; i < 64; ++i)
                acc += efS[e * 64 + i] * W1[i * 128 + t];   // coalesced, L0-hot
            h1S[e * 128 + t] = (__bf16)fmaxf(acc, 0.0f);
        }
    }
    __syncthreads();

    // ---------------- phase 2: WMMA main loop --------------------
    const int lane = t & 31;
    const int lm   = lane & 15;              // N / M-low index
    const int hi   = lane >> 4;              // lane half
    const int m    = (t >> 5) * 16 + lm;     // local edge row for A fragments

    // resident h1 fragments (A 16x32 bf16: lanes 0-15 K {0..7,16..23},
    // lanes 16-31 K {8..15,24..31})
    v16bf h1f[4];
    #pragma unroll
    for (int kb = 0; kb < 4; ++kb) {
        int base = m * 128 + kb * 32 + hi * 8;
        union { v16bf v; int4 q[2]; } u;
        u.q[0] = *(const int4*)&h1S[base];
        u.q[1] = *(const int4*)&h1S[base + 16];
        h1f[kb] = u.v;
    }
    __syncthreads();   // h1 region now free -> becomes TDM buffer 1

    v8f acc[4] = {};   // 4 N-tiles of 16 -> full 64 output columns

    for (int d = 0; d < W_INC; ++d) {
        if (t < 32)
            __builtin_amdgcn_s_wait_tensorcnt(0);  // block d landed in LDS
        __syncthreads();                            // publish buffer to all waves
        if (t < 32 && d + 1 < W_INC)                // stream next block (safe:
            tdm_load_16k((d + 1) & 1 ? lds_b1 : lds_b0,   // consumed 2 iters ago)
                         W2T + (d + 1) * 8192);

        const __bf16* w2cur = (d & 1) ? w2b1 : w2b0;

        // per-lane feats scalar for this d (row m)
        float fs = (float)featsS[m * 80 + d];
        __bf16 fb = (__bf16)fs;
        v16bf fsv;
        #pragma unroll
        for (int i = 0; i < 16; ++i) fsv[i] = fb;

        #pragma unroll
        for (int kb = 0; kb < 4; ++kb) {
            v16bf a = h1f[kb] * fsv;   // P fragment = feats[m,d] * h1[m, kblk]
            #pragma unroll
            for (int nt = 0; nt < 4; ++nt) {
                // B 32x16 bf16: lanes 0-15 hold K {0..7,16..23}, lanes 16-31 K {8..15,24..31}
                int n    = nt * 16 + lm;
                int boff = n * 128 + kb * 32 + hi * 8;
                union { v16bf v; int4 q[2]; } ub;
                ub.q[0] = *(const int4*)&w2cur[boff];
                ub.q[1] = *(const int4*)&w2cur[boff + 16];
                acc[nt] = __builtin_amdgcn_wmma_f32_16x16x32_bf16(
                    false, a, false, ub.v, (short)0, acc[nt], false, false);
            }
        }
    }

    // ---------------- epilogue: residual, b2 term, LayerNorm -----
    const float inv_sqrt_win = 0.11180339887498949f;  // 1/sqrt(80)
    #pragma unroll
    for (int nt = 0; nt < 4; ++nt) {
        #pragma unroll
        for (int r = 0; r < 8; ++r) {
            int e = (t >> 5) * 16 + hi * 8 + r;  // C layout: vgpr r -> rows r/r+8
            int h = nt * 16 + lm;
            efS[e * 64 + h] += acc[nt][r] * inv_sqrt_win;
        }
    }
    __syncthreads();

    // reuse buffer 0 region for b2 (bf16)
    #pragma unroll
    for (int j = 0; j < 40; ++j) {
        int idx = t + 128 * j;                // 5120 elems
        w2b0[idx] = (__bf16)b2[idx];
    }
    __syncthreads();

    // x += (feats @ b2_reshaped) / sqrt(80)
    for (int j = 0; j < 32; ++j) {
        int idx = t + 128 * j;
        int e = idx >> 6, h = idx & 63;
        float bs = 0.f;
        #pragma unroll 8
        for (int d = 0; d < W_INC; ++d)
            bs += (float)featsS[e * 80 + d] * (float)w2b0[d * 64 + h];
        efS[idx] += bs * inv_sqrt_win;
    }
    __syncthreads();

    if (t < 64) {
        float s = 0.f, ss = 0.f;
        #pragma unroll 8
        for (int h = 0; h < 64; ++h) {
            float v = efS[t * 64 + h];
            s += v; ss += v * v;
        }
        float mu  = s * (1.0f / 64.0f);
        float var = ss * (1.0f / 64.0f) - mu * mu;
        muS[t] = mu;
        rsS[t] = rsqrtf(var + 1e-5f);
    }
    __syncthreads();

    #pragma unroll
    for (int j = 0; j < 32; ++j) {
        int idx = t + 128 * j;
        int e = idx >> 6, h = idx & 63;
        out[e0 * 64 + idx] = (efS[idx] - muS[e]) * rsS[e] * gamma_[h] + beta_[h];
    }
}

// ---------------------------------------------------------------------------
extern "C" void kernel_launch(void* const* d_in, const int* in_sizes, int n_in,
                              void* d_out, int out_size, void* d_ws, size_t ws_size,
                              hipStream_t stream) {
    const float* res  = (const float*)d_in[0];   // res_features  N x 56
    const float* ef   = (const float*)d_in[1];   // edge_features E x 64
    const float* esh  = (const float*)d_in[2];   // edge_sh       E x 4
    const float* W1   = (const float*)d_in[3];   // 64 x 128
    const float* b1   = (const float*)d_in[4];   // 128
    const float* W2   = (const float*)d_in[5];   // 128 x 5120
    const float* b2   = (const float*)d_in[6];   // 5120
    const float* gam  = (const float*)d_in[7];   // 64
    const float* bet  = (const float*)d_in[8];   // 64
    const long long* eidx = (const long long*)d_in[9]; // 2 x E int64
    float* out = (float*)d_out;

    __bf16* W2T = (__bf16*)d_ws;                 // 80*64*128 bf16 = 1.31 MB

    w2t_kernel<<<(W_INC * H_DIM * H1_DIM + 255) / 256, 256, 0, stream>>>(W2, W2T);
    edge_main<<<E_EDGES / 64, 128, 0, stream>>>(res, ef, esh, eidx, W1, b1, b2,
                                                gam, bet, W2T, out);
}
